// Transformer_35167192220217
// MI455X (gfx1250) — matrix operand order repaired and checked by reading, not compiled, not run on hardware
//
#include <hip/hip_runtime.h>
#include <hip/hip_bf16.h>

// ---------------------------------------------------------------------------
// Transformer forward for MI455X (gfx1250, wave32, WMMA).
// All GEMMs route through V_WMMA_F32_16X16X32_BF16 with f32 accumulation.
// f32 global data converted to bf16 (RNE, packed via V_PERM_B32) while
// staging tiles through ping-pong LDS buffers. 128xBN block tiles, 32xWN
// wave tiles, register-staged double buffering, float4 global loads,
// b64 LDS stores.
// ---------------------------------------------------------------------------

typedef __attribute__((ext_vector_type(16))) __bf16          v16bf;
typedef __attribute__((ext_vector_type(2)))  __bf16          v2bf;
typedef __attribute__((ext_vector_type(16))) unsigned short  v16u;
typedef __attribute__((ext_vector_type(2)))  unsigned int    uint2v;
typedef __attribute__((ext_vector_type(8)))  float           v8f;

static __device__ inline unsigned short f32_bf16_bits(float f) {
    unsigned u = __float_as_uint(f);
    u += 0x7FFFu + ((u >> 16) & 1u);   // round-to-nearest-even
    return (unsigned short)(u >> 16);
}

// pack two f32 into two bf16 (low = a, high = b), round-to-nearest-even
static __device__ inline unsigned int cvt_pk_bf16(float a, float b) {
#if __has_builtin(__builtin_amdgcn_cvt_pk_bf16_f32)
    v2bf t = __builtin_amdgcn_cvt_pk_bf16_f32(a, b);
    return __builtin_bit_cast(unsigned int, t);
#else
    unsigned ua = __float_as_uint(a);
    unsigned ub = __float_as_uint(b);
    ua += 0x7FFFu + ((ua >> 16) & 1u);
    ub += 0x7FFFu + ((ub >> 16) & 1u);
    // v_perm_b32: sel bytes 0-3 pick from src1 (ua), 4-7 from src0 (ub).
    // result = { ub[31:24], ub[23:16], ua[31:24], ua[23:16] }
    return __builtin_amdgcn_perm(ub, ua, 0x07060302u);
#endif
}

// ---------------------------------------------------------------------------
// Generic tiled GEMM:  C[z] = act( A[z] (MxK) * B[z]^T (NxK) + bias[z] )
//   BT=true : B stored row-major [N, Kdim], leading dim ldb  (weights)
//   BT=false: B stored row-major [Kdim, N], leading dim ldb  (activations)
// Batch z = blockIdx.z decomposed (zo=z/zdiv, zi=z%zdiv) with independent
// outer/inner strides so strided [B,S,H,K]-style layouts batch per-(b,h).
//
// Block: 256 threads = 8 wave32, tile BM=128 x BN=2*WN.
// Waves in 4x2 grid; each wave computes 32 x WN via 2 x (WN/16) WMMA accums.
// K staged 32 wide; next tile loaded to registers while WMMAs run; LDS
// ping-pong buffers give one barrier per K step.
// ---------------------------------------------------------------------------
template <int WN, bool BT, bool RELU, bool BIAS>
__global__ __launch_bounds__(256) void gemm_wmma(
    const float* __restrict__ A,  long long a_off, int lda,
    long long a_zo, long long a_zi,
    const float* __restrict__ Bm, long long b_off, int ldb,
    long long b_zo, long long b_zi,
    const float* __restrict__ bias, long long bias_off, long long bias_zi,
    float* __restrict__ C, long long c_off, int ldc,
    long long c_zo, long long c_zi,
    int M, int N, int Kdim, int zdiv)
{
    constexpr int BM = 128;
    constexpr int BN = 2 * WN;
    constexpr int BK = 32;
    constexpr int LDP = BK + 4;              // row pad: 72B rows, 8B aligned
    constexpr int NACC   = WN / 16;          // accumulators per row-frag
    constexpr int AELEM4 = BM * BK / 1024;   // float4 per thread for A (4)
    constexpr int BELEM4 = BN * BK / 1024;   // float4 per thread for B (4/2)
    constexpr int BELEM  = BN * BK / 256;    // scalar path (BT=false)

    __shared__ unsigned short As[2][BM][LDP];
    __shared__ unsigned short Bs[2][BN][LDP];

    const int z  = blockIdx.z;
    const int zo = z / zdiv;
    const int zi = z % zdiv;

    const float* Ab = A  + a_off + (long long)zo * a_zo + (long long)zi * a_zi;
    const float* Bb = Bm + b_off + (long long)zo * b_zo + (long long)zi * b_zi;
    float*       Cb = C  + c_off + (long long)zo * c_zo + (long long)zi * c_zi;

    const int m0 = blockIdx.y * BM;
    const int n0 = blockIdx.x * BN;

    const int tid  = threadIdx.x;
    const int wave = tid >> 5;              // 0..7
    const int lane = tid & 31;
    const int wm   = (wave >> 1) * 32;      // wave row offset: 0/32/64/96
    const int wn   = (wave & 1) * WN;       // wave col offset: 0/WN

    v8f acc[2][NACC] = {};

    float4 aR[AELEM4];
    float4 bR4[BELEM4];
    float  bReg[BELEM];

    auto loadA = [&](int kk) {
        #pragma unroll
        for (int i = 0; i < AELEM4; ++i) {
            int e = tid * 4 + i * 1024;
            int r = e >> 5, c = e & 31;
            aR[i] = *reinterpret_cast<const float4*>(
                &Ab[(long long)(m0 + r) * lda + (kk + c)]);
        }
    };
    auto loadB = [&](int kk) {
        if constexpr (BT) {
            #pragma unroll
            for (int i = 0; i < BELEM4; ++i) {
                int e = tid * 4 + i * 1024;
                int r = e >> 5, c = e & 31;   // r = n index, c = k index
                const float4* p = reinterpret_cast<const float4*>(
                    &Bb[(long long)(n0 + r) * ldb + (kk + c)]);
                bR4[i] = *p;
                if (kk + BK < Kdim) __builtin_prefetch(p + BK / 4, 0, 3);
            }
        } else {
            #pragma unroll
            for (int i = 0; i < BELEM; ++i) {
                int e = tid + i * 256;
                int r = e >> 5, c = e & 31;
                bReg[i] = Bb[(long long)(kk + c) * ldb + (n0 + r)];
            }
        }
    };
    auto stageLDS = [&](int sb) {
        #pragma unroll
        for (int i = 0; i < AELEM4; ++i) {
            int e = tid * 4 + i * 1024;
            int r = e >> 5, c = e & 31;
            uint2v p;
            p[0] = cvt_pk_bf16(aR[i].x, aR[i].y);
            p[1] = cvt_pk_bf16(aR[i].z, aR[i].w);
            *reinterpret_cast<uint2v*>(&As[sb][r][c]) = p;   // ds_store_b64
        }
        if constexpr (BT) {
            #pragma unroll
            for (int i = 0; i < BELEM4; ++i) {
                int e = tid * 4 + i * 1024;
                int r = e >> 5, c = e & 31;
                uint2v p;
                p[0] = cvt_pk_bf16(bR4[i].x, bR4[i].y);
                p[1] = cvt_pk_bf16(bR4[i].z, bR4[i].w);
                *reinterpret_cast<uint2v*>(&Bs[sb][r][c]) = p;
            }
        } else {
            #pragma unroll
            for (int i = 0; i < BELEM; ++i) {
                int e = tid + i * 256;
                Bs[sb][e >> 5][e & 31] = f32_bf16_bits(bReg[i]);
            }
        }
    };

    // prologue: stage first K tile into buffer 0
    loadA(0);
    loadB(0);
    stageLDS(0);
    __syncthreads();

    const int arow0 = wm + (lane & 15);
    const int khalf = (lane & 16) ? 8 : 0;  // A: lanes16-31 hold K 8..15/24..31
    const int kb    = (lane & 16) ? 8 : 0;  // B: lanes16-31 hold K 8..15/24..31

    int buf = 0;
    for (int k0 = BK; k0 <= Kdim; k0 += BK) {
        const bool has_next = (k0 < Kdim);
        if (has_next) { loadA(k0); loadB(k0); }   // overlap with WMMAs below

        // ---- A fragments (16x32 bf16, ISA 7.12.2 layout) ----------------
        v16bf af[2];
        #pragma unroll
        for (int mi = 0; mi < 2; ++mi) {
            int row = arow0 + mi * 16;
            v16u au;
            #pragma unroll
            for (int j = 0; j < 8; ++j) au[j]     = As[buf][row][khalf + j];
            #pragma unroll
            for (int j = 0; j < 8; ++j) au[8 + j] = As[buf][row][16 + khalf + j];
            af[mi] = __builtin_bit_cast(v16bf, au);
        }
        // ---- B fragments (32x16 each, lane = column) --------------------
        #pragma unroll
        for (int ni = 0; ni < NACC; ++ni) {
            int col = wn + ni * 16 + (lane & 15);
            v16u bu;
            #pragma unroll
            for (int j = 0; j < 16; ++j) bu[j] = Bs[buf][col][kb + j + (j & 8)];
            v16bf bf = __builtin_bit_cast(v16bf, bu);
            #pragma unroll
            for (int mi = 0; mi < 2; ++mi) {
                acc[mi][ni] = __builtin_amdgcn_wmma_f32_16x16x32_bf16(
                    false, af[mi], false, bf, (short)0, acc[mi][ni],
                    false, false);
            }
        }
        if (has_next) stageLDS(buf ^ 1);          // write other buffer
        __syncthreads();
        buf ^= 1;
    }

    // ---- epilogue: C layout, VGPR r -> row r (lanes 0-15) / r+8 ---------
    const int rowhi = (lane & 16) ? 8 : 0;
    const int ncl   = lane & 15;

    #pragma unroll
    for (int ni = 0; ni < NACC; ++ni) {
        int n = n0 + wn + ni * 16 + ncl;
        float bv = 0.0f;
        if (BIAS) bv = bias[bias_off + (long long)zi * bias_zi + n];
        #pragma unroll
        for (int mi = 0; mi < 2; ++mi) {
            #pragma unroll
            for (int r = 0; r < 8; ++r) {
                int m = m0 + wm + mi * 16 + rowhi + r;
                float v = acc[mi][ni][r] + bv;
                if (RELU) v = fmaxf(v, 0.0f);
                Cb[(long long)m * ldc + n] = v;
            }
        }
    }
}

// ---------------------------------------------------------------------------
// x[b,s,:] = emb[tokens[b,s],:] + pe[s,:]
// ---------------------------------------------------------------------------
__global__ __launch_bounds__(256) void embed_kernel(
    const int* __restrict__ tokens, const float* __restrict__ emb,
    const float* __restrict__ pe, float* __restrict__ x,
    int S_, int E_, long long total)
{
    long long i = (long long)blockIdx.x * 256 + threadIdx.x;
    if (i >= total) return;
    long long e  = i % E_;
    long long bs = i / E_;
    long long s  = bs % S_;
    int tok = tokens[bs];
    x[i] = emb[(long long)tok * E_ + e] + pe[s * E_ + e];
}

// ---------------------------------------------------------------------------
// masked softmax over rows of scores [B,H,S,S] (in place)
// mask: tril incl. diag gets -1e10 (faithful to reference)
// ---------------------------------------------------------------------------
__global__ __launch_bounds__(256) void softmax_mask_kernel(
    float* __restrict__ sc, int S_, float scale)
{
    __shared__ float redm[8];
    __shared__ float reds[8];
    long long row  = blockIdx.x;            // over B*H*S
    int       srow = (int)(row % S_);
    float*    p    = sc + row * (long long)S_;
    int       tid  = threadIdx.x;

    float vals[4];
    float mx = -3.4e38f;
    #pragma unroll
    for (int i = 0; i < 4; ++i) {
        int t = tid + i * 256;
        float v = p[t] * scale + ((t <= srow) ? -1e10f : 0.0f);
        vals[i] = v;
        mx = fmaxf(mx, v);
    }
    #pragma unroll
    for (int off = 16; off > 0; off >>= 1) mx = fmaxf(mx, __shfl_xor(mx, off, 32));
    if ((tid & 31) == 0) redm[tid >> 5] = mx;
    __syncthreads();
    float m = redm[0];
    #pragma unroll
    for (int i = 1; i < 8; ++i) m = fmaxf(m, redm[i]);

    float sum = 0.0f;
    #pragma unroll
    for (int i = 0; i < 4; ++i) { vals[i] = __expf(vals[i] - m); sum += vals[i]; }
    #pragma unroll
    for (int off = 16; off > 0; off >>= 1) sum += __shfl_xor(sum, off, 32);
    if ((tid & 31) == 0) reds[tid >> 5] = sum;
    __syncthreads();
    float s = 0.0f;
    #pragma unroll
    for (int i = 0; i < 8; ++i) s += reds[i];
    float inv = 1.0f / s;
    #pragma unroll
    for (int i = 0; i < 4; ++i) p[tid + i * 256] = vals[i] * inv;
}

// ---------------------------------------------------------------------------
// y = LayerNorm(x + o) * w + b      (one block per row of E=1024)
// ---------------------------------------------------------------------------
static __device__ inline float block_sum256(float v, float* red) {
    #pragma unroll
    for (int off = 16; off > 0; off >>= 1) v += __shfl_xor(v, off, 32);
    int tid = threadIdx.x;
    if ((tid & 31) == 0) red[tid >> 5] = v;
    __syncthreads();
    float s = 0.0f;
    #pragma unroll
    for (int i = 0; i < 8; ++i) s += red[i];
    __syncthreads();
    return s;
}

__global__ __launch_bounds__(256) void add_ln_kernel(
    const float* __restrict__ x, const float* __restrict__ o,
    const float* __restrict__ w, const float* __restrict__ b,
    float* __restrict__ y, int E_)
{
    __shared__ float red[8];
    long long row = blockIdx.x;
    const float* xp = x + row * (long long)E_;
    const float* op = o + row * (long long)E_;
    float*       yp = y + row * (long long)E_;
    int tid = threadIdx.x;

    float vals[4];
    float s = 0.0f;
    #pragma unroll
    for (int i = 0; i < 4; ++i) {
        int e = tid + i * 256;
        float v = xp[e] + op[e];
        vals[i] = v;
        s += v;
    }
    float mean = block_sum256(s, red) / (float)E_;

    float vs = 0.0f;
    #pragma unroll
    for (int i = 0; i < 4; ++i) {
        float d = vals[i] - mean;
        vs += d * d;
    }
    float var = block_sum256(vs, red) / (float)E_;
    float inv = rsqrtf(var + 1e-5f);

    #pragma unroll
    for (int i = 0; i < 4; ++i) {
        int e = tid + i * 256;
        yp[e] = (vals[i] - mean) * inv * w[e] + b[e];
    }
}

// ---------------------------------------------------------------------------
// host-side orchestration
// ---------------------------------------------------------------------------
extern "C" void kernel_launch(void* const* d_in, const int* in_sizes, int n_in,
                              void* d_out, int out_size, void* d_ws, size_t ws_size,
                              hipStream_t stream) {
    constexpr int E = 1024, Kk = 64, Dd = 64, S = 1024, Vv = 32000, H = 16,
                  L = 8, Bb = 2;
    constexpr int M = Bb * S;  // 2048 token rows

    const int*   tokens = (const int*)d_in[0];
    const float* emb    = (const float*)d_in[1];
    const float* pe     = (const float*)d_in[2];
    const float* k1w    = (const float*)d_in[3];
    const float* k1b    = (const float*)d_in[4];
    const float* k2w    = (const float*)d_in[5];
    const float* k2b    = (const float*)d_in[6];
    const float* q1w    = (const float*)d_in[7];
    const float* q1b    = (const float*)d_in[8];
    const float* q2w    = (const float*)d_in[9];
    const float* q2b    = (const float*)d_in[10];
    const float* vw     = (const float*)d_in[11];
    const float* vb     = (const float*)d_in[12];
    const float* lnw    = (const float*)d_in[13];
    const float* lnb    = (const float*)d_in[14];
    const float* p1w    = (const float*)d_in[15];
    const float* p1b    = (const float*)d_in[16];
    const float* p2w    = (const float*)d_in[17];
    const float* p2b    = (const float*)d_in[18];
    const float* predw  = (const float*)d_in[19];
    const float* predb  = (const float*)d_in[20];
    float*       out    = (float*)d_out;

    // workspace layout (all f32)
    float* ws   = (float*)d_ws;
    size_t offw = 0;
    auto alloc  = [&](size_t n) { float* p = ws + offw; offw += n; return p; };
    float* X  = alloc((size_t)M * E);          // activations [B,S,E]
    float* K1 = alloc((size_t)M * H * Kk);     // [B,S,H,K]
    float* K2 = alloc((size_t)M * H * Kk);
    float* Q1 = alloc((size_t)M * H * Kk);
    float* Q2 = alloc((size_t)M * H * Kk);
    float* Vh = alloc((size_t)M * H * Dd);     // [B,S,H,D]
    float* SC = alloc((size_t)Bb * H * S * S); // [B,H,S,S]
    float* Oh = alloc((size_t)M * H * Dd);     // [B,S,H,D]
    float* YN = alloc((size_t)M * E);
    float* H1 = alloc((size_t)M * E);

    // x = emb[tokens] + pe
    {
        long long total = (long long)M * E;
        embed_kernel<<<dim3((unsigned)((total + 255) / 256)), 256, 0, stream>>>(
            tokens, emb, pe, X, S, E, total);
    }

    const dim3 blk(256);
    for (int l = 0; l < L; ++l) {
        // k = relu(X @ k1w[l]^T + k1b[l])   -> [B,S,H*K]
        gemm_wmma<64, true, true, true><<<dim3(H * Kk / 128, M / 128, 1), blk, 0, stream>>>(
            X, 0, E, 0, 0,
            k1w, (long long)l * H * Kk * E, E, 0, 0,
            k1b, (long long)l * H * Kk, 0,
            K1, 0, H * Kk, 0, 0, M, H * Kk, E, 1);
        // k = relu(k @ k2w[l,h]^T + k2b[l,h]) per head (N = 64)
        gemm_wmma<32, true, true, true><<<dim3(1, M / 128, H), blk, 0, stream>>>(
            K1, 0, H * Kk, 0, Kk,
            k2w, (long long)l * H * Kk * Kk, Kk, 0, (long long)Kk * Kk,
            k2b, (long long)l * H * Kk, Kk,
            K2, 0, H * Kk, 0, Kk, M, Kk, Kk, H);
        // q path
        gemm_wmma<64, true, true, true><<<dim3(H * Kk / 128, M / 128, 1), blk, 0, stream>>>(
            X, 0, E, 0, 0,
            q1w, (long long)l * H * Kk * E, E, 0, 0,
            q1b, (long long)l * H * Kk, 0,
            Q1, 0, H * Kk, 0, 0, M, H * Kk, E, 1);
        gemm_wmma<32, true, true, true><<<dim3(1, M / 128, H), blk, 0, stream>>>(
            Q1, 0, H * Kk, 0, Kk,
            q2w, (long long)l * H * Kk * Kk, Kk, 0, (long long)Kk * Kk,
            q2b, (long long)l * H * Kk, Kk,
            Q2, 0, H * Kk, 0, Kk, M, Kk, Kk, H);
        // v = X @ vw[l]^T + vb[l]
        gemm_wmma<64, true, false, true><<<dim3(H * Dd / 128, M / 128, 1), blk, 0, stream>>>(
            X, 0, E, 0, 0,
            vw, (long long)l * H * Dd * E, E, 0, 0,
            vb, (long long)l * H * Dd, 0,
            Vh, 0, H * Dd, 0, 0, M, H * Dd, E, 1);
        // scores[b,h] = q[b,h] @ k[b,h]^T     (S x S, Kdim = K)
        gemm_wmma<64, true, false, false><<<dim3(S / 128, S / 128, Bb * H), blk, 0, stream>>>(
            Q2, 0, H * Kk, (long long)S * H * Kk, Kk,
            K2, 0, H * Kk, (long long)S * H * Kk, Kk,
            nullptr, 0, 0,
            SC, 0, S, (long long)H * S * S, (long long)S * S,
            S, S, Kk, H);
        // softmax(scores/sqrt(K) + mask)
        softmax_mask_kernel<<<dim3(Bb * H * S), blk, 0, stream>>>(SC, S, 0.125f);
        // o[b,h] = w[b,h] @ v[b,h]  (S x D, Kdim = S), B not transposed, N=64
        gemm_wmma<32, false, false, false><<<dim3(1, S / 128, Bb * H), blk, 0, stream>>>(
            SC, 0, S, (long long)H * S * S, (long long)S * S,
            Vh, 0, H * Dd, (long long)S * H * Dd, Dd,
            nullptr, 0, 0,
            Oh, 0, H * Dd, (long long)S * H * Dd, Dd,
            S, Dd, S, H);
        // y = LN(x + o) * lnw + lnb
        add_ln_kernel<<<dim3(M), blk, 0, stream>>>(
            X, Oh, lnw + (size_t)l * E, lnb + (size_t)l * E, YN, E);
        // h1 = relu(y @ p1w^T + p1b)
        gemm_wmma<64, true, true, true><<<dim3(E / 128, M / 128, 1), blk, 0, stream>>>(
            YN, 0, E, 0, 0,
            p1w, (long long)l * E * E, E, 0, 0,
            p1b, (long long)l * E, 0,
            H1, 0, E, 0, 0, M, E, E, 1);
        // x = relu(h1 @ p2w^T + p2b)
        gemm_wmma<64, true, true, true><<<dim3(E / 128, M / 128, 1), blk, 0, stream>>>(
            H1, 0, E, 0, 0,
            p2w, (long long)l * E * E, E, 0, 0,
            p2b, (long long)l * E, 0,
            X, 0, E, 0, 0, M, E, E, 1);
    }

    // logits = X @ predw^T + predb  -> d_out [B,S,V]
    gemm_wmma<64, true, false, true><<<dim3(Vv / 128, M / 128, 1), blk, 0, stream>>>(
        X, 0, E, 0, 0,
        predw, 0, E, 0, 0,
        predb, 0, 0,
        out, 0, Vv, 0, 0, M, Vv, E, 1);

    (void)in_sizes; (void)n_in; (void)out_size; (void)ws_size;
}
